// BuDingAttention_67628555043183
// MI455X (gfx1250) — compile-verified, matrix operand order checked by reading, not run on hardware
//
#include <hip/hip_runtime.h>
#include <hip/hip_bf16.h>

typedef __attribute__((ext_vector_type(16))) _Float16 v16h;
typedef __attribute__((ext_vector_type(8)))  _Float16 v8h;
typedef __attribute__((ext_vector_type(8)))  float    v8f;
typedef __attribute__((ext_vector_type(4)))  unsigned int ui32x4;
typedef __attribute__((ext_vector_type(8)))  int      i32x8;
typedef __attribute__((ext_vector_type(4)))  int      i32x4;

#define B_     2
#define S_     2048
#define HID_   2048
#define NH_    32
#define NKV_   8
#define HD_    64
#define GROUPS_ (NH_ / NKV_)
#define SCALE_ 0.125f           // 64^-0.5
#define THETA_ 10000.0f

#define WMMA_F16(a, b, c) \
  __builtin_amdgcn_wmma_f32_16x16x32_f16(false, (a), false, (b), (short)0, (c), false, false)

union Frag16 { v16h v; v8h h[2]; };

// Load a 16x32 f16 A-style fragment (also used for B == rows of B^T).
// lane 0-15: row = r, K {0..7,16..23}; lane 16-31: row = r, K {8..15,24..31}
__device__ __forceinline__ v16h load_frag(const _Float16* base, int ld, int row0, int kb, int lane) {
  const int r = lane & 15, hi = lane >> 4;
  const _Float16* p = base + (long)(row0 + r) * ld + kb + hi * 8;
  Frag16 f;
  f.h[0] = *(const v8h*)(p);
  f.h[1] = *(const v8h*)(p + 16);
  return f.v;
}

// ---------------------------------------------------------------- converts
__global__ void cvt_f32_to_f16(const float* __restrict__ in, _Float16* __restrict__ out, long n) {
  long i = (long)blockIdx.x * blockDim.x + threadIdx.x;
  if (i < n) out[i] = (_Float16)in[i];
}

// ------------------------------------------------- TDM: 2D f16 tile -> LDS
// Builds the D# per CDNA5 ISA and issues TENSOR_LOAD_TO_LDS.
// Loads `rows` x 64-element (f16) tile starting at gptr, global row stride = ld elems.
__device__ __forceinline__ void tdm_load_tile64(const _Float16* gptr, unsigned lds_off,
                                                unsigned rows, unsigned ld) {
#if __has_builtin(__builtin_amdgcn_tensor_load_to_lds)
  const unsigned long long ga = (unsigned long long)(uintptr_t)gptr;
  ui32x4 g0;
  g0[0] = 1u;                                              // count=1, user mode
  g0[1] = lds_off;                                         // lds_addr (bytes)
  g0[2] = (unsigned)(ga & 0xffffffffull);                  // global_addr[31:0]
  g0[3] = (unsigned)((ga >> 32) & 0x1ffffffull) | (2u << 30); // addr[56:32] | type=2
  const unsigned td0 = ld;        // tensor_dim0 (elems)
  const unsigned td1 = rows;      // tensor_dim1 (rows)
  const unsigned tile0 = 64u;     // tile_dim0: 64 f16 = 128B (mult. of 4B)
  const unsigned tile1 = rows;    // tile_dim1
  i32x8 g1;
  g1[0] = (int)(1u << 16);                                      // data_size=2B, no mask
  g1[1] = (int)((td0 & 0xffffu) << 16);                         // dim0[15:0] @ bit48
  g1[2] = (int)(((td0 >> 16) & 0xffffu) | ((td1 & 0xffffu) << 16));
  g1[3] = (int)(((td1 >> 16) & 0xffffu) | (tile0 << 16));       // tile_dim0 @ bit112
  g1[4] = (int)(tile1 & 0xffffu);                               // tile_dim1, tile_dim2=0
  g1[5] = (int)ld;                                              // dim0_stride[31:0]
  g1[6] = 0;                                                    // stride hi, dim1_stride
  g1[7] = 0;
  i32x4 z4 = {0, 0, 0, 0};
#if defined(__clang_major__) && (__clang_major__ >= 23)
  i32x8 z8 = {0, 0, 0, 0, 0, 0, 0, 0};
  __builtin_amdgcn_tensor_load_to_lds(g0, g1, z4, z4, z8, 0);
#else
  __builtin_amdgcn_tensor_load_to_lds(g0, g1, z4, z4, 0);
#endif
#endif
}

// ------------------------------------------------------- WMMA GEMM (x @ W^T + b)
// A: [M,K] f16 row-major, W: [N,K] f16 row-major, C: [M,N] f32.
// Workgroup (256 thr = 8 waves) computes 128x64; wave = 32x32; K blocked by 64.
// A/W K-blocks are staged in LDS by the Tensor Data Mover (wave 0), compute
// waves read fragments with ds_load_b128.
__global__ __launch_bounds__(256)
void gemm_wmma(const _Float16* __restrict__ A, const _Float16* __restrict__ W,
               const float* __restrict__ bias, float* __restrict__ C,
               int M, int N, int K) {
  const int lane = threadIdx.x & 31;
  const int w = threadIdx.x >> 5;
  const int tiles_n = N >> 6;
  const int tm = blockIdx.x / tiles_n, tn = blockIdx.x % tiles_n;
  const int wm = (w >> 1) * 32;          // wave row offset within 128
  const int wn = (w & 1) * 32;           // wave col offset within 64

  __shared__ __align__(16) _Float16 Atile[128 * 64];   // 16 KB
  __shared__ __align__(16) _Float16 Wtile[64 * 64];    // 8 KB

  v8f c00 = {}, c01 = {}, c10 = {}, c11 = {};

  for (int kb = 0; kb < K; kb += 64) {
#if __has_builtin(__builtin_amdgcn_tensor_load_to_lds)
    if (w == 0) {
      tdm_load_tile64(A + (long)tm * 128 * K + kb,
                      (unsigned)(uintptr_t)Atile, 128u, (unsigned)K);
      tdm_load_tile64(W + (long)tn * 64 * K + kb,
                      (unsigned)(uintptr_t)Wtile, 64u, (unsigned)K);
      __builtin_amdgcn_s_wait_tensorcnt(0);
    }
#else
    // cooperative fallback: 256 threads copy both tiles as 16B chunks
    {
      const int t = threadIdx.x;
      for (int i = t; i < 128 * 8; i += 256) {           // A: 1024 v8h chunks
        const int row = i >> 3, ch = i & 7;
        *(v8h*)(Atile + row * 64 + ch * 8) =
            *(const v8h*)(A + ((long)tm * 128 + row) * K + kb + ch * 8);
      }
      for (int i = t; i < 64 * 8; i += 256) {            // W: 512 v8h chunks
        const int row = i >> 3, ch = i & 7;
        *(v8h*)(Wtile + row * 64 + ch * 8) =
            *(const v8h*)(W + ((long)tn * 64 + row) * K + kb + ch * 8);
      }
    }
#endif
    __syncthreads();
#pragma unroll
    for (int kk = 0; kk < 64; kk += 32) {
      v16h a0 = load_frag(Atile, 64, wm, kk, lane);
      v16h a1 = load_frag(Atile, 64, wm + 16, kk, lane);
      v16h b0 = load_frag(Wtile, 64, wn, kk, lane);
      v16h b1 = load_frag(Wtile, 64, wn + 16, kk, lane);
      c00 = WMMA_F16(a0, b0, c00);
      c01 = WMMA_F16(a0, b1, c01);
      c10 = WMMA_F16(a1, b0, c10);
      c11 = WMMA_F16(a1, b1, c11);
    }
    __syncthreads();
  }

  const int r = lane & 15, hi = lane >> 4;
  const int m0 = tm * 128 + wm, n0 = tn * 64 + wn;
  const float bb0 = bias ? bias[n0 + r] : 0.f;
  const float bb1 = bias ? bias[n0 + 16 + r] : 0.f;
#pragma unroll
  for (int i = 0; i < 8; ++i) {
    const long ma = m0 + i + 8 * hi;
    const long mb = m0 + 16 + i + 8 * hi;
    C[ma * N + n0 + r]      = c00[i] + bb0;
    C[ma * N + n0 + 16 + r] = c01[i] + bb1;
    C[mb * N + n0 + r]      = c10[i] + bb0;
    C[mb * N + n0 + 16 + r] = c11[i] + bb1;
  }
}

// ----------------------------------------- RoPE + repack [B,S,H,D] -> [B,H,S,D] f16
__global__ void rope_pack(const float* __restrict__ in, _Float16* __restrict__ out, int nheads) {
  long idx = (long)blockIdx.x * blockDim.x + threadIdx.x;
  const long total = (long)B_ * S_ * nheads * HD_;
  if (idx >= total) return;
  const int d = idx & (HD_ - 1);
  long t = idx >> 6;
  const int h = (int)(t % nheads); t /= nheads;
  const int s = (int)(t % S_);
  const int b = (int)(t / S_);
  const float* row = in + ((long)(b * S_ + s)) * nheads * HD_ + h * HD_;
  const float x = row[d];
  const int j = d & 31;
  const float ang = (float)s * __powf(THETA_, -(float)j * (1.0f / 32.0f));
  float sn, cs; __sincosf(ang, &sn, &cs);
  const float val = (d < 32) ? (x * cs - row[d + 32] * sn)
                             : (x * cs + row[d - 32] * sn);
  out[(((long)b * nheads + h) * S_ + s) * HD_ + d] = (_Float16)val;
}

// ------------------------------------- V: [B,S,KV,D] f32 -> V^T [B,KV,D,S] f16
__global__ void v_transpose(const float* __restrict__ in, _Float16* __restrict__ out) {
  long idx = (long)blockIdx.x * blockDim.x + threadIdx.x;
  const long total = (long)B_ * S_ * NKV_ * HD_;
  if (idx >= total) return;
  const int d = idx & (HD_ - 1);
  long t = idx >> 6;
  const int h = (int)(t % NKV_); t /= NKV_;
  const int s = (int)(t % S_);
  const int b = (int)(t / S_);
  const float x = in[((long)(b * S_ + s)) * (NKV_ * HD_) + h * HD_ + d];
  out[(((long)b * NKV_ + h) * HD_ + d) * S_ + s] = (_Float16)x;
}

// ------------------------------------------------- flash attention (causal, GQA)
// grid: (S/16/4, B*NH), block: 128 (4 waves, one 16-row q-tile each)
__global__ void flash_attn(const _Float16* __restrict__ qh, const _Float16* __restrict__ kh,
                           const _Float16* __restrict__ vT, _Float16* __restrict__ attn) {
  const int lane = threadIdx.x & 31;
  const int w = threadIdx.x >> 5;
  const int qt = blockIdx.x * 4 + w;
  const int bh = blockIdx.y;
  const int b = bh / NH_, h = bh % NH_;
  const int kvh = b * NKV_ + (h / GROUPS_);
  const int q0 = qt * 16;

  __shared__ __align__(16) _Float16 lp[4][16 * 32];
  _Float16* myp = lp[w];

  const _Float16* qbase = qh + (long)bh * S_ * HD_;
  const _Float16* kbase = kh + (long)kvh * S_ * HD_;
  const _Float16* vbase = vT + (long)kvh * HD_ * S_;

  const v16h qa0 = load_frag(qbase, HD_, q0, 0, lane);
  const v16h qa1 = load_frag(qbase, HD_, q0, 32, lane);

  v8f o0 = {}, o1 = {}, o2 = {}, o3 = {};
  float mrow[8], lrow[8];
#pragma unroll
  for (int i = 0; i < 8; ++i) { mrow[i] = -1e30f; lrow[i] = 0.f; }

  const int r = lane & 15, hi = lane >> 4;
  const int nkt = (q0 + 47) >> 5;   // 32-key tiles covering keys 0..q0+15

  for (int kt = 0; kt < nkt; ++kt) {
    const int ks = kt * 32;
    // S = Q K^T (16x32 of scores), K dim = HD = 64 -> 2 wmma per 16-col subtile
    v8f s0 = {}, s1 = {};
    {
      v16h b0 = load_frag(kbase, HD_, ks, 0, lane);
      s0 = WMMA_F16(qa0, b0, s0);
      v16h b1 = load_frag(kbase, HD_, ks, 32, lane);
      s0 = WMMA_F16(qa1, b1, s0);
      v16h b2 = load_frag(kbase, HD_, ks + 16, 0, lane);
      s1 = WMMA_F16(qa0, b2, s1);
      v16h b3 = load_frag(kbase, HD_, ks + 16, 32, lane);
      s1 = WMMA_F16(qa1, b3, s1);
    }
    // online softmax across the 32 keys of this tile
#pragma unroll
    for (int i = 0; i < 8; ++i) {
      const int qp = q0 + i + 8 * hi;
      float v0 = s0[i] * SCALE_;
      float v1 = s1[i] * SCALE_;
      if (ks + r > qp)      v0 = -1e30f;
      if (ks + 16 + r > qp) v1 = -1e30f;
      float mx = fmaxf(v0, v1);
      for (int off = 8; off > 0; off >>= 1) mx = fmaxf(mx, __shfl_xor(mx, off, 32));
      const float nm = fmaxf(mrow[i], mx);
      const float p0 = __expf(v0 - nm);
      const float p1 = __expf(v1 - nm);
      float rs = p0 + p1;
      for (int off = 8; off > 0; off >>= 1) rs += __shfl_xor(rs, off, 32);
      const float alpha = __expf(mrow[i] - nm);
      lrow[i] = lrow[i] * alpha + rs;
      mrow[i] = nm;
      o0[i] *= alpha; o1[i] *= alpha; o2[i] *= alpha; o3[i] *= alpha;
      myp[(i + 8 * hi) * 32 + r]      = (_Float16)p0;
      myp[(i + 8 * hi) * 32 + 16 + r] = (_Float16)p1;
    }
    // P (16x32, LDS, A-fragment) x V (32x64 via V^T rows, contiguous)
    const v16h pa = load_frag(myp, 32, 0, 0, lane);
    v16h vb0 = load_frag(vbase, S_, 0, ks, lane);
    o0 = WMMA_F16(pa, vb0, o0);
    v16h vb1 = load_frag(vbase, S_, 16, ks, lane);
    o1 = WMMA_F16(pa, vb1, o1);
    v16h vb2 = load_frag(vbase, S_, 32, ks, lane);
    o2 = WMMA_F16(pa, vb2, o2);
    v16h vb3 = load_frag(vbase, S_, 48, ks, lane);
    o3 = WMMA_F16(pa, vb3, o3);
  }

  // normalize and write back to [B,S, NH*HD] f16 for the O-projection
#pragma unroll
  for (int i = 0; i < 8; ++i) {
    const long row = (long)(b * S_ + q0 + i + 8 * hi) * (NH_ * HD_) + h * HD_;
    const float inv = 1.0f / lrow[i];
    attn[row + r]      = (_Float16)(o0[i] * inv);
    attn[row + 16 + r] = (_Float16)(o1[i] * inv);
    attn[row + 32 + r] = (_Float16)(o2[i] * inv);
    attn[row + 48 + r] = (_Float16)(o3[i] * inv);
  }
}

// ================================================================= launcher
extern "C" void kernel_launch(void* const* d_in, const int* in_sizes, int n_in,
                              void* d_out, int out_size, void* d_ws, size_t ws_size,
                              hipStream_t stream) {
  const float* hs  = (const float*)d_in[0];
  const float* q_w = (const float*)d_in[1];
  const float* q_b = (const float*)d_in[2];
  const float* k_w = (const float*)d_in[3];
  const float* k_b = (const float*)d_in[4];
  const float* v_w = (const float*)d_in[5];
  const float* v_b = (const float*)d_in[6];
  const float* o_w = (const float*)d_in[7];

  const long M = (long)B_ * S_;                 // 4096
  char* ws = (char*)d_ws;
  size_t off = 0;
  auto take = [&](size_t bytes) { char* p = ws + off; off += (bytes + 255) & ~(size_t)255; return p; };

  _Float16* Xh  = (_Float16*)take(M * HID_ * 2);                    // hidden f16
  _Float16* Wqh = (_Float16*)take((size_t)NH_ * HD_ * HID_ * 2);
  _Float16* Wkh = (_Float16*)take((size_t)NKV_ * HD_ * HID_ * 2);
  _Float16* Wvh = (_Float16*)take((size_t)NKV_ * HD_ * HID_ * 2);
  _Float16* Woh = (_Float16*)take((size_t)HID_ * NH_ * HD_ * 2);
  float*    qf  = (float*)take(M * NH_ * HD_ * 4);
  float*    kf  = (float*)take(M * NKV_ * HD_ * 4);
  float*    vf  = (float*)take(M * NKV_ * HD_ * 4);
  _Float16* qr  = (_Float16*)take(M * NH_ * HD_ * 2);               // [B,NH,S,D]
  _Float16* kr  = (_Float16*)take(M * NKV_ * HD_ * 2);              // [B,NKV,S,D]
  _Float16* vT  = (_Float16*)take(M * NKV_ * HD_ * 2);              // [B,NKV,D,S]
  _Float16* at  = (_Float16*)take(M * NH_ * HD_ * 2);               // attn out f16

  auto cvt = [&](const float* src, _Float16* dst, long n) {
    cvt_f32_to_f16<<<(int)((n + 255) / 256), 256, 0, stream>>>(src, dst, n);
  };
  cvt(hs,  Xh,  M * HID_);
  cvt(q_w, Wqh, (long)NH_ * HD_ * HID_);
  cvt(k_w, Wkh, (long)NKV_ * HD_ * HID_);
  cvt(v_w, Wvh, (long)NKV_ * HD_ * HID_);
  cvt(o_w, Woh, (long)HID_ * NH_ * HD_);

  auto gemm = [&](const _Float16* A, const _Float16* W, const float* bias, float* C,
                  int m, int n, int k) {
    const int blocks = (m / 128) * (n / 64);
    gemm_wmma<<<blocks, 256, 0, stream>>>(A, W, bias, C, m, n, k);
  };
  gemm(Xh, Wqh, q_b, qf, (int)M, NH_ * HD_, HID_);   // Q proj
  gemm(Xh, Wkh, k_b, kf, (int)M, NKV_ * HD_, HID_);  // K proj
  gemm(Xh, Wvh, v_b, vf, (int)M, NKV_ * HD_, HID_);  // V proj

  {
    const long nq = M * NH_ * HD_;
    rope_pack<<<(int)((nq + 255) / 256), 256, 0, stream>>>(qf, qr, NH_);
    const long nk = M * NKV_ * HD_;
    rope_pack<<<(int)((nk + 255) / 256), 256, 0, stream>>>(kf, kr, NKV_);
    v_transpose<<<(int)((nk + 255) / 256), 256, 0, stream>>>(vf, vT);
  }

  {
    dim3 grid(S_ / 16 / 4, B_ * NH_);
    flash_attn<<<grid, 128, 0, stream>>>(qr, kr, vT, at);
  }

  // O projection -> f32 output
  gemm(at, Woh, nullptr, (float*)d_out, (int)M, HID_, NH_ * HD_);
  (void)in_sizes; (void)n_in; (void)out_size; (void)ws_size;
}
